// AttentionLayer_32693291057192
// MI455X (gfx1250) — compile-verified, matrix operand order checked by reading, not run on hardware
//
#include <hip/hip_runtime.h>
#include <hip/hip_bf16.h>

// ---------------------------------------------------------------------------
// Transformer-XL attention for MI455X (gfx1250, wave32, WMMA bf16).
//
// B=2, L=1024, E=1024, H=16, DH=64, MEM=1024, LMEM=64, KV=2112, TM=1088.
// ~52 GFLOP total, working set ~60MB (L2-resident at 192MB) -> compute bound:
// every matmul runs on v_wmma_f32_16x16x32_bf16 (f32 accumulate).
// All activations/weights are staged to bf16 ONCE (elementwise kernels), so
// GEMM inner loops are pure global_load_b128 + v_wmma (no conversion VALU).
// Attention is flash-style (no (B,H,L,KV) logits tensor); the Transformer-XL
// relative shift is folded into a diagonal-band WMMA + cross-lane shuffle.
// ---------------------------------------------------------------------------

typedef __attribute__((ext_vector_type(16))) __bf16          v16bf;
typedef __attribute__((ext_vector_type(8)))  float           v8f;
typedef __attribute__((ext_vector_type(16))) unsigned short  v16u;
typedef __attribute__((ext_vector_type(8)))  unsigned short  v8u;

#define BB   2
#define LL   1024
#define EE   1024
#define HH   16
#define DHD  64
#define KVK  2112
#define TMM  1088   // mem + lmem

__device__ __forceinline__ unsigned short f2bf(float f) {
  unsigned u = __builtin_bit_cast(unsigned, f);
  u += 0x7FFFu + ((u >> 16) & 1u);              // round-to-nearest-even
  return (unsigned short)(u >> 16);
}

__device__ __forceinline__ v8f vzero() {
  v8f z;
#pragma unroll
  for (int i = 0; i < 8; ++i) z[i] = 0.0f;
  return z;
}

__device__ __forceinline__ v8f wmma_bf16(v16bf a, v16bf b, v8f c) {
  return __builtin_amdgcn_wmma_f32_16x16x32_bf16(false, a, false, b,
                                                 (short)0, c, false, false);
}

// A/B fragment K-map (wave32, 16x16x32 bf16):
//   lanes 0-15  hold K in {0..7, 16..23}   (kA = 0)
//   lanes 16-31 hold K in {8..15, 24..31}  (kA = 8)
// -> fragment = two contiguous 8-element (16B) runs at p[0..7] and p[16..23],
//    where p already includes the +kA lane offset.
__device__ __forceinline__ v16bf frag_bf16(const unsigned short* p) {
  v8u lo = *reinterpret_cast<const v8u*>(p);
  v8u hi = *reinterpret_cast<const v8u*>(p + 16);
  v16u u;
#pragma unroll
  for (int i = 0; i < 8; ++i) { u[i] = lo[i]; u[i + 8] = hi[i]; }
  return __builtin_bit_cast(v16bf, u);
}

// -------------------- prep kernels --------------------------------------- //

// W (K=1024 x N=1024 f32, row-major) -> WT bf16 (N x K row-major)
__global__ __launch_bounds__(256) void k_transp(const float* __restrict__ W,
                                               unsigned short* __restrict__ WT) {
  int i = blockIdx.x * 256 + threadIdx.x;      // grid sized exactly: 1M elems
  int n = i >> 10, k = i & 1023;
  WT[i] = f2bf(W[k * 1024 + n]);
}

__global__ __launch_bounds__(256) void k_cvt(const float* __restrict__ src,
                                             unsigned short* __restrict__ dst,
                                             int n) {
  int i = blockIdx.x * 256 + threadIdx.x;
  if (i < n) dst[i] = f2bf(src[i]);
}

// Build kv_input = concat(lmem, mem, queries) as bf16 (B*KV x E row-major).
// Queries rows double as the A-matrix for the Q projection (rows TMM..KV-1).
__global__ __launch_bounds__(256) void k_kvb(const float* __restrict__ queries,
                                             const float* __restrict__ mem,
                                             const float* __restrict__ lmem,
                                             unsigned short* __restrict__ kvb) {
  int i = blockIdx.x * 256 + threadIdx.x;      // grid exact: BB*KVK*EE elems
  int r = i >> 10, k = i & 1023;
  int b = r / KVK, s = r - b * KVK;
  const float* src =
      (s < 64)   ? lmem    + ((size_t)b * 64 + s) * EE
    : (s < TMM)  ? mem     + ((size_t)b * 1024 + (s - 64)) * EE
                 : queries + ((size_t)b * 1024 + (s - TMM)) * EE;
  kvb[i] = f2bf(src[k]);
}

// -------------------- QKV projection (WMMA GEMM) ------------------------- //
// A = kv_input bf16 (B*KV x E). One 16x16 tile per wave, K-loop of 32.
// mode 0: Q  = queries-rows @ WqT, out (B,H,L,64)   bf16
// mode 1: K  = kv_input     @ WkT, out (B,H,KV,64)  bf16
// mode 2: V  = kv_input     @ WvT, out (B,H,64,KV)  bf16 (transposed for P@V)
__global__ __launch_bounds__(256) void k_proj(const unsigned short* __restrict__ kvb,
                                              const unsigned short* __restrict__ WT,
                                              const float* __restrict__ bias,
                                              unsigned short* __restrict__ out,
                                              int mode, int S) {
  int wid = threadIdx.x >> 5, lane = threadIdx.x & 31;
  int ln = lane & 15; bool hi = lane >= 16; int kA = hi ? 8 : 0;
  int t = blockIdx.x * 8 + wid;                 // one 16x16 tile per wave
  int tM = (2 * S) >> 4;
  if (t >= tM * 64) return;                     // wave-uniform guard
  int m0 = (t >> 6) << 4;
  int n0 = (t & 63) << 4;

  int rowA = m0 + ln;
  int kvrow;
  if (mode == 0) kvrow = (rowA >> 10) * KVK + TMM + (rowA & 1023);
  else           kvrow = rowA;
  const unsigned short* aptr = kvb + (size_t)kvrow * EE;
  const unsigned short* bptr = WT + (size_t)(n0 + ln) * EE;

  v8f c = vzero();
#pragma unroll 4
  for (int k0 = 0; k0 < EE; k0 += 32) {
    __builtin_prefetch(bptr + ((k0 + 256) & 1023), 0, 3);  // global_prefetch_b8
    v16bf a = frag_bf16(aptr + k0 + kA);
    v16bf b = frag_bf16(bptr + k0 + kA);
    c = wmma_bf16(a, b, c);
  }

  int col = n0 + ln, h = col >> 6, d = col & 63;
  float bv = bias[col];
#pragma unroll
  for (int g = 0; g < 8; ++g) {
    int row = m0 + g + (hi ? 8 : 0);
    int b = row / S, s = row - b * S;
    unsigned short o = f2bf(c[g] + bv);
    size_t idx;
    if (mode == 0)      idx = (((size_t)(b * HH + h)) * LL  + s) * DHD + d;
    else if (mode == 1) idx = (((size_t)(b * HH + h)) * KVK + s) * DHD + d;
    else                idx = (((size_t)(b * HH + h)) * DHD + d) * KVK + s;
    out[idx] = o;
  }
}

// -------------------- flash attention w/ rel-shift ----------------------- //
// One wave per (b, h, 16-row q tile). Streams KV in 32-col steps.
// pos term: shifted[q,m] = q[q] . pe[m + L-1 - q]; per 16x16 tile this is a
// 31-wide band of Q@PE^T -> 3 overlapping WMMA tiles + diagonal extraction
// via lane shuffles (C layout: VGPR=row, lane=col).
__global__ __launch_bounds__(256) void k_attn(const unsigned short* __restrict__ qb,
                                              const unsigned short* __restrict__ kb,
                                              const unsigned short* __restrict__ vtb,
                                              const unsigned short* __restrict__ peb,
                                              unsigned short* __restrict__ attn_out) {
  __shared__ __align__(16) unsigned short ldsP[8][16 * 32];  // per-wave P tile

  int wid = threadIdx.x >> 5, lane = threadIdx.x & 31;
  int ln = lane & 15; bool hi = lane >= 16; int kA = hi ? 8 : 0;
  int t = blockIdx.x * 8 + wid;                 // 0..2047
  int qt = t & 63, h = (t >> 6) & 15, b = t >> 10;
  int q0 = qt << 4;

  const unsigned short* qrow = qb + (((size_t)(b * HH + h)) * LL + q0 + ln) * DHD;
  v16bf qa0 = frag_bf16(qrow + kA);             // DH 0..31
  v16bf qa1 = frag_bf16(qrow + 32 + kA);        // DH 32..63

  const unsigned short* kbase  = kb  + ((size_t)(b * HH + h)) * KVK * DHD;
  const unsigned short* vbase  = vtb + ((size_t)(b * HH + h)) * DHD * KVK;
  const unsigned short* pebase = peb + (size_t)h * KVK * DHD;

  v8f o[4] = {vzero(), vzero(), vzero(), vzero()};
  float mrun[8], srun[8];
#pragma unroll
  for (int g = 0; g < 8; ++g) { mrun[g] = -1.0e30f; srun[g] = 0.0f; }

  int mEnd = q0 + 16 + TMM;                     // last possibly-unmasked col + 1
  mEnd = (mEnd + 31) & ~31;
  if (mEnd > KVK) mEnd = KVK;

  for (int m0 = 0; m0 < mEnd; m0 += 32) {
    // --- positional band: Q @ PE^T for j0 .. j0+47 (3 x 16-col tiles) ---
    int j0 = m0 + 1008 - q0;                    // = m0 + (L-1) - q0 - 15
    v8f T[3];
#pragma unroll
    for (int jt = 0; jt < 3; ++jt) {
      int j = j0 + jt * 16 + ln;
      if (j > KVK - 1) j = KVK - 1;             // OOB == masked; clamp is safe
      const unsigned short* pr = pebase + (size_t)j * DHD;
      v8f acc = vzero();
      acc = wmma_bf16(qa0, frag_bf16(pr + kA), acc);
      acc = wmma_bf16(qa1, frag_bf16(pr + 32 + kA), acc);
      T[jt] = acc;
    }

    // --- logits for two 16-col tiles: dots + shifted pos, scale, mask ----
    v8f Sfrag[2];
#pragma unroll
    for (int ct = 0; ct < 2; ++ct) {
      int mt = m0 + ct * 16;
      const unsigned short* kr = kbase + (size_t)(mt + ln) * DHD;
      v8f acc = vzero();
      acc = wmma_bf16(qa0, frag_bf16(kr + kA), acc);
      acc = wmma_bf16(qa1, frag_bf16(kr + 32 + kA), acc);
#pragma unroll
      for (int g = 0; g < 8; ++g) {
        int r = g + (hi ? 8 : 0);
        int bc = ln + 15 - r;                   // band col 0..30
        int srcLane = (bc & 15) + (hi ? 16 : 0);
        float p0 = __shfl(T[ct][g], srcLane, 32);
        float p1 = __shfl(T[ct + 1][g], srcLane, 32);
        float pv = (bc < 16) ? p0 : p1;
        float s = (acc[g] + pv) * 0.125f;       // 1/sqrt(DH)
        if (mt + ln > q0 + r + TMM) s = -3.0e38f;  // causal+mem mask
        acc[g] = s;
      }
      Sfrag[ct] = acc;
    }

    // --- online softmax (rows live in VGPR index; reduce across 16 lanes) -
#pragma unroll
    for (int g = 0; g < 8; ++g) {
      float rm = fmaxf(Sfrag[0][g], Sfrag[1][g]);
#pragma unroll
      for (int off = 1; off < 16; off <<= 1) rm = fmaxf(rm, __shfl_xor(rm, off, 32));
      float mnew = fmaxf(mrun[g], rm);
      float alpha = __expf(mrun[g] - mnew);
      mrun[g] = mnew;
      float e0 = __expf(Sfrag[0][g] - mnew);
      float e1 = __expf(Sfrag[1][g] - mnew);
      Sfrag[0][g] = e0; Sfrag[1][g] = e1;
      float rs = e0 + e1;
#pragma unroll
      for (int off = 1; off < 16; off <<= 1) rs += __shfl_xor(rs, off, 32);
      srun[g] = srun[g] * alpha + rs;
#pragma unroll
      for (int dt = 0; dt < 4; ++dt) o[dt][g] *= alpha;
    }

    // --- P: C-layout -> A-layout via per-wave LDS tile -------------------
    unsigned short* lp = &ldsP[wid][0];
#pragma unroll
    for (int ct = 0; ct < 2; ++ct)
#pragma unroll
      for (int g = 0; g < 8; ++g) {
        int r = g + (hi ? 8 : 0);
        lp[r * 32 + ct * 16 + ln] = f2bf(Sfrag[ct][g]);
      }
    const unsigned short* lpr = lp + ln * 32;
    v16bf pa = frag_bf16(lpr + kA);             // P tile 16x32 as A fragment

    // --- O += P @ V  (V^T rows are contiguous in kv) ---------------------
#pragma unroll
    for (int dt = 0; dt < 4; ++dt) {
      const unsigned short* vr = vbase + (size_t)(dt * 16 + ln) * KVK + m0;
      o[dt] = wmma_bf16(pa, frag_bf16(vr + kA), o[dt]);
    }
  }

  // --- normalize and emit (B, L, H*DH) bf16 for the output projection ----
#pragma unroll
  for (int dt = 0; dt < 4; ++dt)
#pragma unroll
    for (int g = 0; g < 8; ++g) {
      int r = g + (hi ? 8 : 0);
      float val = o[dt][g] / srun[g];
      attn_out[((size_t)(b * LL + q0 + r)) * (HH * DHD) + h * DHD + dt * 16 + ln] =
          f2bf(val);
    }
}

// -------------------- output projection (WMMA GEMM, f32 out) ------------- //
__global__ __launch_bounds__(256) void k_out(const unsigned short* __restrict__ A,
                                             const unsigned short* __restrict__ WT,
                                             const float* __restrict__ bias,
                                             float* __restrict__ out) {
  int wid = threadIdx.x >> 5, lane = threadIdx.x & 31;
  int ln = lane & 15; bool hi = lane >= 16; int kA = hi ? 8 : 0;
  int t = blockIdx.x * 8 + wid;                 // 8192 tiles: 128 x 64
  int m0 = (t >> 6) << 4;
  int n0 = (t & 63) << 4;

  const unsigned short* aptr = A + (size_t)(m0 + ln) * EE;
  const unsigned short* bptr = WT + (size_t)(n0 + ln) * EE;

  v8f c = vzero();
#pragma unroll 4
  for (int k0 = 0; k0 < EE; k0 += 32) {
    __builtin_prefetch(bptr + ((k0 + 256) & 1023), 0, 3);  // global_prefetch_b8
    c = wmma_bf16(frag_bf16(aptr + k0 + kA), frag_bf16(bptr + k0 + kA), c);
  }

  int col = n0 + ln;
  float bv = bias[col];
#pragma unroll
  for (int g = 0; g < 8; ++g) {
    int row = m0 + g + (hi ? 8 : 0);
    out[(size_t)row * EE + col] = c[g] + bv;
  }
}

// -------------------- launch --------------------------------------------- //
extern "C" void kernel_launch(void* const* d_in, const int* in_sizes, int n_in,
                              void* d_out, int out_size, void* d_ws, size_t ws_size,
                              hipStream_t stream) {
  (void)in_sizes; (void)n_in; (void)out_size; (void)ws_size;

  const float* queries = (const float*)d_in[0];
  // d_in[1] keys, d_in[2] values, d_in[3] attn_mask: unused by the reference
  const float* mem     = (const float*)d_in[4];
  const float* lmem    = (const float*)d_in[5];
  const float* pos_emb = (const float*)d_in[6];
  const float* Wq = (const float*)d_in[7];
  const float* bq = (const float*)d_in[8];
  const float* Wk = (const float*)d_in[9];
  const float* bk = (const float*)d_in[10];
  const float* Wv = (const float*)d_in[11];
  const float* bv = (const float*)d_in[12];
  const float* Wo = (const float*)d_in[13];
  const float* bo = (const float*)d_in[14];
  float* out = (float*)d_out;

  // workspace layout (bf16 staging, 256B aligned), ~48 MB total
  char* ws = (char*)d_ws;
  size_t off = 0;
  auto take = [&](size_t bytes) {
    size_t r = off; off = (off + bytes + 255) & ~(size_t)255; return r;
  };
  unsigned short* kvb   = (unsigned short*)(ws + take((size_t)BB * KVK * EE * 2));
  unsigned short* qb    = (unsigned short*)(ws + take((size_t)BB * HH * LL  * DHD * 2));
  unsigned short* kbuf  = (unsigned short*)(ws + take((size_t)BB * HH * KVK * DHD * 2));
  unsigned short* vT    = (unsigned short*)(ws + take((size_t)BB * HH * DHD * KVK * 2));
  unsigned short* peb   = (unsigned short*)(ws + take((size_t)HH * KVK * DHD * 2));
  unsigned short* WqT   = (unsigned short*)(ws + take((size_t)EE * EE * 2));
  unsigned short* WkT   = (unsigned short*)(ws + take((size_t)EE * EE * 2));
  unsigned short* WvT   = (unsigned short*)(ws + take((size_t)EE * EE * 2));
  unsigned short* WoT   = (unsigned short*)(ws + take((size_t)EE * EE * 2));
  unsigned short* attnb = (unsigned short*)(ws + take((size_t)BB * LL * HH * DHD * 2));

  // 1) one-shot bf16 staging: weights (transposed), pos_emb, kv_input concat
  k_transp<<<4096, 256, 0, stream>>>(Wq, WqT);
  k_transp<<<4096, 256, 0, stream>>>(Wk, WkT);
  k_transp<<<4096, 256, 0, stream>>>(Wv, WvT);
  k_transp<<<4096, 256, 0, stream>>>(Wo, WoT);
  {
    int n = HH * KVK * DHD;
    k_cvt<<<(n + 255) / 256, 256, 0, stream>>>(pos_emb, peb, n);
  }
  k_kvb<<<(BB * KVK * EE) / 256, 256, 0, stream>>>(queries, mem, lmem, kvb);

  // 2) projections (pure b128-load + WMMA inner loops)
  k_proj<<<1024, 256, 0, stream>>>(kvb, WqT, bq, qb,   0, LL);
  k_proj<<<2112, 256, 0, stream>>>(kvb, WkT, bk, kbuf, 1, KVK);
  k_proj<<<2112, 256, 0, stream>>>(kvb, WvT, bv, vT,   2, KVK);

  // 3) flash attention with relative shift (one wave per (b,h,q-tile))
  k_attn<<<256, 256, 0, stream>>>(qb, kbuf, vT, peb, attnb);

  // 4) output projection -> f32
  k_out<<<1024, 256, 0, stream>>>(attnb, WoT, bo, out);
}
